// RecurrentGCN_75076028334105
// MI455X (gfx1250) — compile-verified, matrix operand order checked by reading, not run on hardware
//
#include <hip/hip_runtime.h>
#include <math.h>

// ---------- types for WMMA ----------
typedef __attribute__((ext_vector_type(16))) __bf16          v16bf;
typedef __attribute__((ext_vector_type(8)))  float           v8f;

__device__ __forceinline__ float sigmoidf_(float x) {
    return 1.0f / (1.0f + __expf(-x));
}

// Load one lane's A-fragment chunk for V_WMMA_F32_16X16X32_BF16.
// base points at row_start + half*8 floats; lane needs floats [0..7] and [16..23].
// Native bf16 casts -> v_cvt_pk_bf16_f32 pairs.
__device__ __forceinline__ v16bf load_a_rowchunk(const float* __restrict__ base) {
    const float4* p = reinterpret_cast<const float4*>(base);
    float4 f0 = p[0], f1 = p[1], f2 = p[4], f3 = p[5];
    v16bf a;
    a[0]=(__bf16)f0.x;  a[1]=(__bf16)f0.y;  a[2]=(__bf16)f0.z;  a[3]=(__bf16)f0.w;
    a[4]=(__bf16)f1.x;  a[5]=(__bf16)f1.y;  a[6]=(__bf16)f1.z;  a[7]=(__bf16)f1.w;
    a[8]=(__bf16)f2.x;  a[9]=(__bf16)f2.y;  a[10]=(__bf16)f2.z; a[11]=(__bf16)f2.w;
    a[12]=(__bf16)f3.x; a[13]=(__bf16)f3.y; a[14]=(__bf16)f3.z; a[15]=(__bf16)f3.w;
    return a;
}

// Pre-swizzled B-fragment: lane's 16 bf16 are contiguous -> 2x ds_load_b128.
__device__ __forceinline__ v16bf load_frag(const __bf16* p) {
    return *reinterpret_cast<const v16bf*>(p);
}

// Map fragment element j (0..15) + lane-half to the B-matrix K index.
__device__ __forceinline__ int frag_k(int half, int j) {
    return half * 8 + ((j < 8) ? j : (16 + (j - 8)));
}

// ---------- elementwise / graph-prep kernels ----------
__global__ void k_zero_f4(float4* __restrict__ p, int n4) {
    int i = blockIdx.x * blockDim.x + threadIdx.x;
    if (i < n4) p[i] = make_float4(0.f, 0.f, 0.f, 0.f);
}

__global__ void k_negcopy_f4(float4* __restrict__ d, const float4* __restrict__ s, int n4) {
    int i = blockIdx.x * blockDim.x + threadIdx.x;
    if (i < n4) {
        float4 v = s[i];
        d[i] = make_float4(-v.x, -v.y, -v.z, -v.w);
    }
}

__global__ void k_deg(float* __restrict__ deg, const int* __restrict__ ei,
                      const float* __restrict__ ew, int E) {
    int e = blockIdx.x * blockDim.x + threadIdx.x;
    if (e >= E) return;
    int r = ei[e], c = ei[E + e];
    if (r != c) atomicAdd(deg + r, ew[e]);
}

__global__ void k_dis(float* __restrict__ deg, int N) {
    int n = blockIdx.x * blockDim.x + threadIdx.x;
    if (n >= N) return;
    float d = deg[n];
    deg[n] = (d > 0.f) ? rsqrtf(fmaxf(d, 1e-12f)) : 0.f;   // in-place -> dis
}

__global__ void k_wn(float* __restrict__ wn, const int* __restrict__ ei,
                     const float* __restrict__ ew, const float* __restrict__ dis, int E) {
    int e = blockIdx.x * blockDim.x + threadIdx.x;
    if (e >= E) return;
    int r = ei[e], c = ei[E + e];
    float w = (r != c) ? ew[e] : 0.f;
    wn[e] = dis[r] * w * dis[c];
}

// ---------- temporal conv 1 : (12,N,1) -> (N,10,32), gated + relu ----------
__global__ void k_tconv1(const float* __restrict__ x, const float* __restrict__ tw,
                         const float* __restrict__ tb, float* __restrict__ h1, int N) {
    int idx = blockIdx.x * blockDim.x + threadIdx.x;          // N*10*32 threads
    if (idx >= N * 10 * 32) return;
    int c  = idx & 31;
    int rt = idx >> 5;          // n*10 + t
    int n  = rt / 10;
    int t  = rt - n * 10;
    float P = tb[c], Q = tb[32 + c], R = tb[64 + c];
#pragma unroll
    for (int k = 0; k < 3; ++k) {
        float xv = x[(size_t)(t + k) * N + n];
        P += xv * tw[c * 3 + k];
        Q += xv * tw[96 + c * 3 + k];
        R += xv * tw[192 + c * 3 + k];
    }
    h1[(size_t)rt * 32 + c] = fmaxf(P * sigmoidf_(Q) + R, 0.f);
}

// ---------- sparse propagation: dst[row] += coef * wn * src[col], wave per edge ----------
__global__ void k_prop(float* __restrict__ dst, const float* __restrict__ src,
                       const int* __restrict__ ei, const float* __restrict__ wn,
                       float coef, int E) {
    int gtid = blockIdx.x * blockDim.x + threadIdx.x;
    int e = gtid >> 5, lane = gtid & 31;
    if (e >= E) return;
    int row = ei[e], col = ei[E + e];
    float w = coef * wn[e];
    const float* s = src + (size_t)col * 320;
    float*       d = dst + (size_t)row * 320;
#pragma unroll
    for (int i = 0; i < 10; ++i)
        atomicAdd(d + i * 32 + lane, w * s[i * 32 + lane]);
}

// ---------- Chebyshev mixing: out = relu(sum_k Tx_k @ W_k + b), WMMA ----------
// sF layout: [k][ntile][lane][16] bf16, pre-swizzled so each lane's fragment is
// a contiguous 32B LDS read.
__global__ void k_cheb_mm(const float* __restrict__ tx0, const float* __restrict__ tx1,
                          const float* __restrict__ tx2, const float* __restrict__ cw,
                          const float* __restrict__ cb, float* __restrict__ h2, int ntiles) {
    __shared__ __align__(32) __bf16 sF[3 * 2 * 32 * 16];   // 6 KB
    for (int i = threadIdx.x; i < 3072; i += blockDim.x) {
        int k    = i >> 10;
        int rem  = i & 1023;
        int nt   = rem >> 9;
        int rem2 = rem & 511;
        int lane = rem2 >> 4;
        int j    = rem2 & 15;
        int half = lane >> 4;
        int col  = nt * 16 + (lane & 15);
        int K    = frag_k(half, j);
        sF[i] = (__bf16)cw[k * 1024 + K * 32 + col];
    }
    __syncthreads();

    int lane = threadIdx.x & 31;
    int tile = blockIdx.x * (blockDim.x >> 5) + (threadIdx.x >> 5);
    if (tile >= ntiles) return;

    int half = lane >> 4, mrow = lane & 15, c0 = lane & 15;
    v8f acc0 = {0.f,0.f,0.f,0.f,0.f,0.f,0.f,0.f};
    v8f acc1 = acc0;
    int arow = tile * 16 + mrow;
#pragma unroll
    for (int k = 0; k < 3; ++k) {
        const float* tx = (k == 0) ? tx0 : ((k == 1) ? tx1 : tx2);
        v16bf a  = load_a_rowchunk(tx + (size_t)arow * 32 + half * 8);
        v16bf b0 = load_frag(sF + ((k * 2 + 0) * 32 + lane) * 16);
        acc0 = __builtin_amdgcn_wmma_f32_16x16x32_bf16(false, a, false, b0,
                                                       (short)0, acc0, false, false);
        v16bf b1 = load_frag(sF + ((k * 2 + 1) * 32 + lane) * 16);
        acc1 = __builtin_amdgcn_wmma_f32_16x16x32_bf16(false, a, false, b1,
                                                       (short)0, acc1, false, false);
    }
    float bb0 = cb[c0], bb1 = cb[c0 + 16];
#pragma unroll
    for (int v = 0; v < 8; ++v) {
        int row = tile * 16 + v + 8 * half;         // C/D layout: vgpr v, halves
        h2[(size_t)row * 32 + c0]      = fmaxf(acc0[v] + bb0, 0.f);
        h2[(size_t)row * 32 + 16 + c0] = fmaxf(acc1[v] + bb1, 0.f);
    }
}

// ---------- temporal conv 2 : (N,10,32) -> (N,8,32), gated + relu, WMMA K=96 ----------
// sF layout: [gate][kc][ntile][lane][16] bf16, pre-swizzled fragments.
__global__ void k_tconv2(const float* __restrict__ h2, const float* __restrict__ w2,
                         const float* __restrict__ b2, float* __restrict__ h3, int ntiles) {
    __shared__ __align__(32) __bf16 sF[3 * 3 * 2 * 32 * 16];   // 18 KB
    for (int i = threadIdx.x; i < 9216; i += blockDim.x) {
        int g    = i / 3072;
        int rem  = i - g * 3072;
        int kc   = rem >> 10;
        int rem2 = rem & 1023;
        int nt   = rem2 >> 9;
        int rem3 = rem2 & 511;
        int lane = rem3 >> 4;
        int j    = rem3 & 15;
        int half = lane >> 4;
        int col  = nt * 16 + (lane & 15);          // co
        int ci   = frag_k(half, j);                // K within this kc chunk == ci
        sF[i] = (__bf16)w2[g * 3072 + col * 96 + ci * 3 + kc];
    }
    __syncthreads();

    int lane = threadIdx.x & 31;
    int tile = blockIdx.x * (blockDim.x >> 5) + (threadIdx.x >> 5);
    if (tile >= ntiles) return;

    int half = lane >> 4, mrow = lane & 15, c0 = lane & 15;
    int r = tile * 16 + mrow;
    int n = r >> 3, t = r & 7;
    const float* rowb = h2 + (size_t)n * 320 + t * 32;   // 96 contiguous floats

    v16bf A[3];
#pragma unroll
    for (int kc = 0; kc < 3; ++kc)
        A[kc] = load_a_rowchunk(rowb + kc * 32 + half * 8);

    v8f acc[3][2];
#pragma unroll
    for (int g = 0; g < 3; ++g)
#pragma unroll
        for (int nt = 0; nt < 2; ++nt)
            acc[g][nt] = (v8f){0.f,0.f,0.f,0.f,0.f,0.f,0.f,0.f};

#pragma unroll
    for (int g = 0; g < 3; ++g) {
#pragma unroll
        for (int kc = 0; kc < 3; ++kc) {
            const __bf16* fb = sF + ((g * 3 + kc) * 2) * 512 + lane * 16;
            v16bf b0 = load_frag(fb);
            acc[g][0] = __builtin_amdgcn_wmma_f32_16x16x32_bf16(false, A[kc], false, b0,
                                                                (short)0, acc[g][0], false, false);
            v16bf b1 = load_frag(fb + 512);
            acc[g][1] = __builtin_amdgcn_wmma_f32_16x16x32_bf16(false, A[kc], false, b1,
                                                                (short)0, acc[g][1], false, false);
        }
    }
#pragma unroll
    for (int nt = 0; nt < 2; ++nt) {
        int col = nt * 16 + c0;
        float bp = b2[col], bq = b2[32 + col], br = b2[64 + col];
#pragma unroll
        for (int v = 0; v < 8; ++v) {
            int row = tile * 16 + v + 8 * half;
            float P = acc[0][nt][v] + bp;
            float Q = acc[1][nt][v] + bq;
            float R = acc[2][nt][v] + br;
            h3[(size_t)row * 32 + col] = fmaxf(P * sigmoidf_(Q) + R, 0.f);
        }
    }
}

// ---------- fused BN + relu + time-mean + linear head : wave per node ----------
__global__ void k_head(const float* __restrict__ h3, const float* __restrict__ gma,
                       const float* __restrict__ bta, const float* __restrict__ lw,
                       const float* __restrict__ lb, float* __restrict__ out, int N) {
    int gtid = blockIdx.x * blockDim.x + threadIdx.x;
    int n = gtid >> 5, lane = gtid & 31;
    if (n >= N) return;

    float v[8], s = 0.f, sq = 0.f;
#pragma unroll
    for (int t = 0; t < 8; ++t) {
        v[t] = h3[(size_t)n * 256 + t * 32 + lane];
        s  += v[t];
        sq += v[t] * v[t];
    }
#pragma unroll
    for (int m = 16; m >= 1; m >>= 1) {
        s  += __shfl_xor(s,  m, 32);
        sq += __shfl_xor(sq, m, 32);
    }
    float mu  = s * (1.f / 256.f);
    float var = sq * (1.f / 256.f) - mu * mu;
    float rs  = rsqrtf(var + 1e-5f);
    float scale = gma[n] * rs;
    float shift = bta[n] - mu * scale;

    float mean_c = 0.f;
#pragma unroll
    for (int t = 0; t < 8; ++t)
        mean_c += fmaxf(v[t] * scale + shift, 0.f);
    mean_c *= 0.125f;

#pragma unroll
    for (int j = 0; j < 4; ++j) {
        float p = mean_c * lw[j * 32 + lane];
#pragma unroll
        for (int m = 16; m >= 1; m >>= 1) p += __shfl_xor(p, m, 32);
        if (lane == 0) out[(size_t)n * 4 + j] = p + lb[j];
    }
}

// ---------- host launcher ----------
extern "C" void kernel_launch(void* const* d_in, const int* in_sizes, int n_in,
                              void* d_out, int out_size, void* d_ws, size_t ws_size,
                              hipStream_t stream) {
    const float* x      = (const float*)d_in[0];
    const int*   ei     = (const int*)  d_in[1];
    const float* ew     = (const float*)d_in[2];
    const float* tc1w   = (const float*)d_in[3];
    const float* tc1b   = (const float*)d_in[4];
    const float* chebw  = (const float*)d_in[5];
    const float* chebb  = (const float*)d_in[6];
    const float* tc2w   = (const float*)d_in[7];
    const float* tc2b   = (const float*)d_in[8];
    const float* bng    = (const float*)d_in[9];
    const float* bnb    = (const float*)d_in[10];
    const float* lw     = (const float*)d_in[11];
    const float* lb     = (const float*)d_in[12];
    float* out = (float*)d_out;

    const int N = in_sizes[9];          // 20000
    const int E = in_sizes[1] / 2;      // 640000
    const size_t NT = (size_t)N * 320;  // (N,10,32) state

    // workspace carve (all 16B aligned: N, E, NT are multiples of 4)
    float* deg = (float*)d_ws;
    float* wn  = deg + N;
    float* h1  = wn + E;        // Tx0
    float* tx1 = h1 + NT;
    float* tx2 = tx1 + NT;
    float* h2  = tx2 + NT;
    float* h3  = h2 + NT;       // (N,8,32)

    const int B = 256;
    auto blk = [](long long w, int b) { return (int)((w + b - 1) / b); };

    // graph normalization
    k_zero_f4<<<blk(N / 4, B), B, 0, stream>>>((float4*)deg, N / 4);
    k_deg    <<<blk(E, B), B, 0, stream>>>(deg, ei, ew, E);
    k_dis    <<<blk(N, B), B, 0, stream>>>(deg, N);
    k_wn     <<<blk(E, B), B, 0, stream>>>(wn, ei, ew, deg, E);

    // temporal conv 1 -> h1 (Tx0)
    k_tconv1 <<<blk((long long)N * 320, B), B, 0, stream>>>(x, tc1w, tc1b, h1, N);

    // Tx1 = -A_hat @ Tx0
    k_zero_f4<<<blk((long long)(NT / 4), B), B, 0, stream>>>((float4*)tx1, (int)(NT / 4));
    k_prop   <<<blk((long long)E * 32, B), B, 0, stream>>>(tx1, h1, ei, wn, -1.0f, E);

    // Tx2 = -2*A_hat @ Tx1 - Tx0
    k_negcopy_f4<<<blk((long long)(NT / 4), B), B, 0, stream>>>((float4*)tx2, (const float4*)h1,
                                                               (int)(NT / 4));
    k_prop   <<<blk((long long)E * 32, B), B, 0, stream>>>(tx2, tx1, ei, wn, -2.0f, E);

    // Cheb mixing (WMMA) -> h2, fused bias+relu
    int tiles1 = (N * 10) / 16;                       // rows divisible by 16 (N mult of 8)
    k_cheb_mm<<<blk((long long)tiles1 * 32, B), B, 0, stream>>>(h1, tx1, tx2, chebw, chebb,
                                                                h2, tiles1);

    // temporal conv 2 (WMMA, K=96) -> h3
    int tiles2 = (N * 8) / 16;
    k_tconv2 <<<blk((long long)tiles2 * 32, B), B, 0, stream>>>(h2, tc2w, tc2b, h3, tiles2);

    // BN + relu + time-mean + linear head -> out (N,4)
    k_head   <<<blk((long long)N * 32, B), B, 0, stream>>>(h3, bng, bnb, lw, lb, out, N);
}
